// InterestEvolutionLayer_6090263626301
// MI455X (gfx1250) — compile-verified, matrix operand order checked by reading, not run on hardware
//
#include <hip/hip_runtime.h>
#include <hip/hip_bf16.h>

// ---------------------------------------------------------------------------
// InterestEvolutionLayer (bilinear attention + AUGRU scan) for gfx1250.
// B=4096, S=200, H=128, E=64.
// GEMMs: v_wmma_f32_16x16x32_bf16. Recurrent state h stays in VGPRs.
// x_t is double-buffered in LDS via global_load_async_to_lds_b128 (ASYNCcnt).
// Gate nonlinearities use v_exp_f32 + v_rcp_f32 (no IEEE-div fixup chains).
// ---------------------------------------------------------------------------

typedef __bf16 bf16_t;
typedef bf16_t v16bf __attribute__((ext_vector_type(16)));
typedef float  v8f   __attribute__((ext_vector_type(8)));

#define B_TOT  4096
#define S_LEN  200
#define H_DIM  128
#define E_DIM  64
#define ROWS   64      // batch rows per workgroup
#define NWAVES 8       // 8 waves * 32 lanes = 256 threads

// float -> bf16 (round to nearest even, NaN edge ignored for this workload)
static __device__ __forceinline__ unsigned short f2bf_bits(float f) {
    union { float f; unsigned u; } c; c.f = f;
    unsigned r = c.u + 0x7fffu + ((c.u >> 16) & 1u);
    return (unsigned short)(r >> 16);
}
static __device__ __forceinline__ bf16_t f2bf(float f) {
    union { unsigned short s; bf16_t b; } o; o.s = f2bf_bits(f);
    return o.b;
}

// sigmoid via v_exp_f32 + v_rcp_f32; exact at the extremes:
//   exp(-x)->inf => rcp->0 ; exp(-x)->0 => 1
static __device__ __forceinline__ float fast_sigmoid(float x) {
    return __builtin_amdgcn_rcpf(1.f + __expf(-x));
}
// tanh(x) = 1 - 2/(exp(2x)+1); exp->inf => 1, exp->0 => -1
static __device__ __forceinline__ float fast_tanh(float x) {
    return 1.f - 2.f * __builtin_amdgcn_rcpf(1.f + __expf(2.f * x));
}

typedef union { uint4 q[2]; v16bf v; } Frag;

// Load one 16x32 (A) or 32x16 (B, from transposed weights) bf16 fragment.
// buf is row-major with row stride H_DIM elements; per ISA layout each lane
// holds 16 values: K = kb..kb+7 and kb+16..kb+23 with kb = 8*(lane>=16).
static __device__ __forceinline__ v16bf ld_frag(const bf16_t* __restrict__ buf,
                                                int row0, int ktile, int lane) {
    int r  = row0 + (lane & 15);
    int kb = ktile * 32 + ((lane >> 4) << 3);
    const bf16_t* p = buf + r * H_DIM + kb;
    Frag f;
    f.q[0] = *(const uint4*)(p);
    f.q[1] = *(const uint4*)(p + 16);
    return f.v;
}

static __device__ __forceinline__ v8f wmma_bf16(v16bf a, v16bf b, v8f c) {
    return __builtin_amdgcn_wmma_f32_16x16x32_bf16(
        /*neg_a=*/false, a, /*neg_b=*/false, b,
        /*c_mod=*/(short)0, c, /*reuse_a=*/false, /*reuse_b=*/false);
}

// ---------------------------------------------------------------------------
// Kernel 1: bilinear attention logits + softmax over S.  One WG per batch row.
// Also (optionally) emits a bf16 copy of interest_states for the async scan.
// ---------------------------------------------------------------------------
__global__ __launch_bounds__(256) void attn_kernel(
    const float* __restrict__ xin,      // [B,S,H]
    const float* __restrict__ target,   // [B,E]
    const float* __restrict__ Wbil,     // [H,E]
    float* __restrict__ att,            // [B,S]
    bf16_t* __restrict__ xbf)           // [B,S,H] bf16 copy (may be null)
{
    __shared__ float vbuf[H_DIM];
    __shared__ float lbuf[S_LEN];
    __shared__ float red[NWAVES];

    const int b    = blockIdx.x;
    const int tid  = threadIdx.x;
    const int lane = tid & 31;
    const int w    = tid >> 5;

    if (tid < H_DIM) {
        float acc = 0.f;
        const float* wrow = Wbil + tid * E_DIM;
        const float* tv   = target + (size_t)b * E_DIM;
#pragma unroll 8
        for (int e = 0; e < E_DIM; ++e) acc += wrow[e] * tv[e];
        vbuf[tid] = acc;
    }
    __syncthreads();

    for (int s = w; s < S_LEN; s += NWAVES) {
        const float* xr = xin + ((size_t)b * S_LEN + s) * H_DIM;
        bf16_t* xw = xbf ? (xbf + ((size_t)b * S_LEN + s) * H_DIM) : (bf16_t*)0;
        float p = 0.f;
#pragma unroll
        for (int k = 0; k < H_DIM / 32; ++k) {
            float xv = xr[lane + 32 * k];
            p += xv * vbuf[lane + 32 * k];
            if (xw) xw[lane + 32 * k] = f2bf(xv);
        }
#pragma unroll
        for (int off = 16; off > 0; off >>= 1) p += __shfl_xor(p, off, 32);
        if (lane == 0) lbuf[s] = p;
    }
    __syncthreads();

    float x = (tid < S_LEN) ? lbuf[tid] : -1e30f;
    float m = x;
#pragma unroll
    for (int off = 16; off > 0; off >>= 1) m = fmaxf(m, __shfl_xor(m, off, 32));
    if (lane == 0) red[w] = m;
    __syncthreads();
    m = red[0];
#pragma unroll
    for (int i = 1; i < NWAVES; ++i) m = fmaxf(m, red[i]);

    float e = (tid < S_LEN) ? __expf(x - m) : 0.f;
    float sum = e;
#pragma unroll
    for (int off = 16; off > 0; off >>= 1) sum += __shfl_xor(sum, off, 32);
    __syncthreads();
    if (lane == 0) red[w] = sum;
    __syncthreads();
    sum = 0.f;
#pragma unroll
    for (int i = 0; i < NWAVES; ++i) sum += red[i];

    if (tid < S_LEN) att[(size_t)b * S_LEN + tid] = e * __builtin_amdgcn_rcpf(sum);
}

// ---------------------------------------------------------------------------
// Kernel 2: pack the six HxH weight matrices to bf16, transposed (WT[n][k]).
// Order: xu,hu,xr,hr,xc,hc.
// ---------------------------------------------------------------------------
__global__ __launch_bounds__(256) void pack_kernel(
    const float* __restrict__ w0, const float* __restrict__ w1,
    const float* __restrict__ w2, const float* __restrict__ w3,
    const float* __restrict__ w4, const float* __restrict__ w5,
    bf16_t* __restrict__ out)
{
    int idx = blockIdx.x * 256 + threadIdx.x;     // 6 * 128 * 128 total
    int mi  = idx >> 14;
    int r   = idx & 16383;
    int n   = r >> 7;
    int k   = r & 127;
    const float* srcs[6] = { w0, w1, w2, w3, w4, w5 };
    out[mi * (H_DIM * H_DIM) + n * H_DIM + k] = f2bf(srcs[mi][k * H_DIM + n]);
}

// ---------------------------------------------------------------------------
// Kernel 3: fused AUGRU scan.  64 batch rows per WG, 8 waves; wave w owns
// output columns [16w, 16w+16).  Weight B-fragments live in registers for the
// entire 200-step scan; h is f32 in registers with a bf16 shadow in LDS.
// ASYNC=true: x_t comes from the bf16 copy via global_load_async_to_lds_b128
// into a double buffer (fetch s+1 while computing s).
// ---------------------------------------------------------------------------
template <bool ASYNC>
__global__ __launch_bounds__(256) void gru_kernel(
    const float*  __restrict__ xin,   // [B,S,H] f32 (fallback path)
    const bf16_t* __restrict__ xbf,   // [B,S,H] bf16 (async path)
    const float*  __restrict__ att,   // [B,S]
    const bf16_t* __restrict__ wT,    // 6 * H*H bf16, transposed
    const float*  __restrict__ bu,
    const float*  __restrict__ br,
    const float*  __restrict__ bc,
    float* __restrict__ out)          // [B,H]
{
    __shared__ bf16_t xbuf[2][ROWS * H_DIM];
    __shared__ bf16_t hbuf[ROWS * H_DIM];
    __shared__ bf16_t rhbuf[ROWS * H_DIM];
    __shared__ float  abuf[ROWS];

    const int tid  = threadIdx.x;
    const int lane = tid & 31;
    const int w    = tid >> 5;
    const int b0   = blockIdx.x * ROWS;
    const int col  = w * 16 + (lane & 15);   // this lane's output column

    // --- resident weight fragments (B operands), 192 VGPRs/lane ---
    v16bf wxu[4], whu[4], wxr[4], whr[4], wxc[4], whc[4];
#pragma unroll
    for (int kt = 0; kt < 4; ++kt) {
        wxu[kt] = ld_frag(wT + 0 * H_DIM * H_DIM, w * 16, kt, lane);
        whu[kt] = ld_frag(wT + 1 * H_DIM * H_DIM, w * 16, kt, lane);
        wxr[kt] = ld_frag(wT + 2 * H_DIM * H_DIM, w * 16, kt, lane);
        whr[kt] = ld_frag(wT + 3 * H_DIM * H_DIM, w * 16, kt, lane);
        wxc[kt] = ld_frag(wT + 4 * H_DIM * H_DIM, w * 16, kt, lane);
        whc[kt] = ld_frag(wT + 5 * H_DIM * H_DIM, w * 16, kt, lane);
    }
    const float bub = bu[col], brb = br[col], bcb = bc[col];

    // h = 0 (f32 registers, WMMA C/D layout) and bf16 shadow in LDS
    v8f h[4];
#pragma unroll
    for (int m = 0; m < 4; ++m)
#pragma unroll
        for (int j = 0; j < 8; ++j) h[m][j] = 0.f;
    for (int i = tid; i < (ROWS * H_DIM) / 8; i += 256)
        ((uint4*)hbuf)[i] = make_uint4(0u, 0u, 0u, 0u);

    if (ASYNC) {
        // prime buffer 0 with x_0: 1024 16B chunks, 4 per thread (4/wave issue)
#pragma unroll
        for (int j = 0; j < 4; ++j) {
            int c   = j * 256 + tid;
            int row = c >> 4;               // 16 chunks per 128-elem row
            int kc  = (c & 15) << 3;        // bf16 element offset
            const bf16_t* g = xbf + ((size_t)(b0 + row) * S_LEN + 0) * H_DIM + kc;
            unsigned lds = (unsigned)(unsigned long long)(const void*)&xbuf[0][c << 3];
            asm volatile("global_load_async_to_lds_b128 %0, %1, off"
                         :: "v"(lds), "v"((unsigned long long)g) : "memory");
        }
    }
    __syncthreads();

    for (int s = 0; s < S_LEN; ++s) {
        const int cur = ASYNC ? (s & 1) : 0;

        if (ASYNC) {
            // ---- issue async fetch of x_{s+1}; release buffer for step s ----
            if (s + 1 < S_LEN) {
                const int nb = (s + 1) & 1;
#pragma unroll
                for (int j = 0; j < 4; ++j) {
                    int c   = j * 256 + tid;
                    int row = c >> 4;
                    int kc  = (c & 15) << 3;
                    const bf16_t* g =
                        xbf + ((size_t)(b0 + row) * S_LEN + (s + 1)) * H_DIM + kc;
                    unsigned lds =
                        (unsigned)(unsigned long long)(const void*)&xbuf[nb][c << 3];
                    asm volatile("global_load_async_to_lds_b128 %0, %1, off"
                                 :: "v"(lds), "v"((unsigned long long)g) : "memory");
                }
                // async loads complete in order: <=4 outstanding means step-s
                // chunks have landed in LDS.
                asm volatile("s_wait_asynccnt 0x4" ::: "memory");
            } else {
                asm volatile("s_wait_asynccnt 0x0" ::: "memory");
            }
            if (tid < ROWS)
                abuf[tid] = att[(size_t)(b0 + tid) * S_LEN + s];
        } else {
            // ---- fallback: synchronous f32 load + convert into xbuf[0] ----
            int row = tid >> 2;
            int seg = tid & 3;
            const float* src =
                xin + ((size_t)(b0 + row) * S_LEN + s) * H_DIM + seg * 32;
            bf16_t* dst = &xbuf[0][row * H_DIM + seg * 32];
#pragma unroll
            for (int j = 0; j < 8; ++j) {
                float4 f = *(const float4*)(src + j * 4);
                ushort4 pk;
                pk.x = f2bf_bits(f.x); pk.y = f2bf_bits(f.y);
                pk.z = f2bf_bits(f.z); pk.w = f2bf_bits(f.w);
                *(ushort4*)(dst + j * 4) = pk;
            }
            if (tid < ROWS) {
                abuf[tid] = att[(size_t)(b0 + tid) * S_LEN + s];
                if (s + 1 < S_LEN)
                    __builtin_prefetch(
                        xin + ((size_t)(b0 + tid) * S_LEN + (s + 1)) * H_DIM, 0, 0);
            }
        }
        __syncthreads();

        const bf16_t* xcur = &xbuf[cur][0];

        // ---- phase A: u = sigmoid(xWxu + hWhu + bu), r = sigmoid(...), rh ----
        v8f u[4];
#pragma unroll
        for (int m = 0; m < 4; ++m) {
            v8f ua, ra;
#pragma unroll
            for (int j = 0; j < 8; ++j) { ua[j] = bub; ra[j] = brb; }
#pragma unroll
            for (int kt = 0; kt < 4; ++kt) {
                v16bf xa = ld_frag(xcur, m * 16, kt, lane);
                v16bf ha = ld_frag(hbuf, m * 16, kt, lane);
                ua = wmma_bf16(xa, wxu[kt], ua);
                ua = wmma_bf16(ha, whu[kt], ua);
                ra = wmma_bf16(xa, wxr[kt], ra);
                ra = wmma_bf16(ha, whr[kt], ra);
            }
#pragma unroll
            for (int j = 0; j < 8; ++j) {
                u[m][j] = fast_sigmoid(ua[j]);
                float r = fast_sigmoid(ra[j]);
                int row = m * 16 + j + ((lane >> 4) << 3);
                rhbuf[row * H_DIM + col] = f2bf(r * h[m][j]);
            }
        }
        __syncthreads();

        // ---- phase B: c = tanh(xWxc + (r*h)Whc + bc); AUGRU update ----
#pragma unroll
        for (int m = 0; m < 4; ++m) {
            v8f ca;
#pragma unroll
            for (int j = 0; j < 8; ++j) ca[j] = bcb;
#pragma unroll
            for (int kt = 0; kt < 4; ++kt) {
                v16bf xa  = ld_frag(xcur,  m * 16, kt, lane);
                v16bf rha = ld_frag(rhbuf, m * 16, kt, lane);
                ca = wmma_bf16(xa,  wxc[kt], ca);
                ca = wmma_bf16(rha, whc[kt], ca);
            }
#pragma unroll
            for (int j = 0; j < 8; ++j) {
                int row  = m * 16 + j + ((lane >> 4) << 3);
                float c  = fast_tanh(ca[j]);
                float au = abuf[row] * u[m][j];
                float hn = (1.f - au) * h[m][j] + au * c;
                h[m][j]  = hn;
                hbuf[row * H_DIM + col] = f2bf(hn);
            }
        }
        __syncthreads();   // protect xbuf/rhbuf/hbuf before next step
    }

    // ---- write h_final ----
#pragma unroll
    for (int m = 0; m < 4; ++m)
#pragma unroll
        for (int j = 0; j < 8; ++j) {
            int row = m * 16 + j + ((lane >> 4) << 3);
            out[(size_t)(b0 + row) * H_DIM + col] = h[m][j];
        }
}

// ---------------------------------------------------------------------------
// Launch: attention (+ bf16 x copy) -> weight pack -> fused AUGRU scan.
// Workspace: att [B*S] f32 (3.28 MB) | packed weights (196 KB) | xbf (210 MB).
// If d_ws is too small for xbf, fall back to the synchronous-f32 scan.
// ---------------------------------------------------------------------------
extern "C" void kernel_launch(void* const* d_in, const int* in_sizes, int n_in,
                              void* d_out, int out_size, void* d_ws, size_t ws_size,
                              hipStream_t stream) {
    const float* xin    = (const float*)d_in[0];   // interest_states [B,S,H]
    const float* target = (const float*)d_in[1];   // target_item     [B,E]
    const float* Wbil   = (const float*)d_in[2];   // W_bil           [H,E]
    const float* Wxu    = (const float*)d_in[3];
    const float* Whu    = (const float*)d_in[4];
    const float* bu     = (const float*)d_in[5];
    const float* Wxr    = (const float*)d_in[6];
    const float* Whr    = (const float*)d_in[7];
    const float* br     = (const float*)d_in[8];
    const float* Wxc    = (const float*)d_in[9];
    const float* Whc    = (const float*)d_in[10];
    const float* bc     = (const float*)d_in[11];

    float* out = (float*)d_out;

    const size_t attB = (size_t)B_TOT * S_LEN * sizeof(float);
    const size_t wB   = (size_t)6 * H_DIM * H_DIM * sizeof(bf16_t);
    const size_t xbfB = (size_t)B_TOT * S_LEN * H_DIM * sizeof(bf16_t);
    const bool use_async = ws_size >= attB + wB + xbfB;

    float*  att = (float*)d_ws;
    bf16_t* wT  = (bf16_t*)((char*)d_ws + attB);
    bf16_t* xbf = use_async ? (bf16_t*)((char*)d_ws + attB + wB) : (bf16_t*)0;

    attn_kernel<<<B_TOT, 256, 0, stream>>>(xin, target, Wbil, att, xbf);
    pack_kernel<<<(6 * H_DIM * H_DIM) / 256, 256, 0, stream>>>(
        Wxu, Whu, Wxr, Whr, Wxc, Whc, wT);

    if (use_async)
        gru_kernel<true><<<B_TOT / ROWS, 256, 0, stream>>>(
            xin, xbf, att, wT, bu, br, bc, out);
    else
        gru_kernel<false><<<B_TOT / ROWS, 256, 0, stream>>>(
            xin, xbf, att, wT, bu, br, bc, out);
}
